// WordByWordSlideSum_23347442221553
// MI455X (gfx1250) — matrix operand order repaired and checked by reading, not compiled, not run on hardware
//
#include <hip/hip_runtime.h>
#include <hip/hip_bf16.h>

typedef __attribute__((ext_vector_type(16))) __bf16 v16bf;
typedef __attribute__((ext_vector_type(8)))  __bf16 v8bf;
typedef __attribute__((ext_vector_type(8)))  float  v8f;

#define B_      64
#define S_      4096
#define A_      4
#define L_      64
#define D_      300
#define KP      320      // padded K (multiple of 32)
#define W_      60
#define STRIDE_ 5
#define NW_     808
#define NCHUNK  8
#define NWT     101      // windows per chunk (8*101 = 808)
#define NTILE   35       // 16-row tiles per chunk (5*101 + 55 = 560 = 35*16)
#define RB      80       // circular score-row buffer depth (> 74 live span)
#define COLS    128      // columns (a,l) handled per workgroup (half of 256)

static_assert(NCHUNK * NWT == NW_, "chunking");
static_assert(NTILE * 16 == NWT * STRIDE_ + W_ - STRIDE_, "rows per chunk");

// ---------------------------------------------------------------------------
// Kernel 1: normalize l_a rows over D, emit bf16 [B*256][KP] (zero-padded K).
// One wave per row; 8 rows per 256-thread block.
// ---------------------------------------------------------------------------
__global__ __launch_bounds__(256) void k_norm_a(const float* __restrict__ la,
                                                __bf16* __restrict__ wsa) {
  const int lane = threadIdx.x & 31;
  const int wv   = threadIdx.x >> 5;
  const int row  = blockIdx.x * 8 + wv;            // 0 .. B_*256-1
  const float* src = la + (size_t)row * D_;
  float v[10];
  float ss = 0.f;
#pragma unroll
  for (int i = 0; i < 10; ++i) {
    const int k = lane + 32 * i;
    const float x = (k < D_) ? src[k] : 0.f;
    v[i] = x;
    ss += x * x;
  }
#pragma unroll
  for (int off = 16; off >= 1; off >>= 1) ss += __shfl_xor(ss, off, 32);
  const float inv = rsqrtf(ss + 1e-6f);
  __bf16* dst = wsa + (size_t)row * KP;
#pragma unroll
  for (int i = 0; i < 10; ++i) {
    const int k = lane + 32 * i;
    dst[k] = (__bf16)(v[i] * inv);                 // k>=300 writes 0
  }
}

// ---------------------------------------------------------------------------
// Kernel 2: fused normalize(l_s) -> bf16 WMMA scores -> gaussian sliding-max
//           -> masked mean over L -> running max over windows.
// Grid: (b, chunk, colHalf). 256 threads = 8 waves; wave w owns 16 N-columns.
// ---------------------------------------------------------------------------
__global__ __launch_bounds__(256) void k_main(const float* __restrict__ ls,
                                              const __bf16* __restrict__ wsa,
                                              const float* __restrict__ gwin,
                                              float* __restrict__ partials) {
  __shared__ __align__(16) float  s_sc[RB * COLS];   // 40 KB score ring
  __shared__ __align__(16) __bf16 s_a[16 * KP];      // 10 KB A staging (bf16)
  __shared__ float s_redp[256];
  __shared__ float s_redc[256];
  __shared__ float s_gw[W_];
  __shared__ float s_wavg[4];

  const int tid   = threadIdx.x;
  const int lane  = tid & 31;
  const int wv    = tid >> 5;          // 0..7 : N-tile of this wave
  const int b     = blockIdx.x;        // 0..63
  const int chunk = blockIdx.y;        // 0..7
  const int half  = blockIdx.z;        // 0..1 : which 128 of the 256 cols

  if (tid < W_) s_gw[tid] = gwin[tid];

  // ---- Preload B fragments for this wave's 16 columns, all K, into VGPRs.
  // B layout (32x16 bf16): N = lane&15, K = 16*(lane>=16) + elem.
  const int gcol = half * COLS + wv * 16 + (lane & 15);
  const int koff = (lane >= 16) ? 16 : 0;
  v16bf Bf[10];
#pragma unroll
  for (int j = 0; j < 10; ++j) {
    const __bf16* p = wsa + ((size_t)(b * 256 + gcol)) * KP + 32 * j + koff;
    const v8bf lo = *(const v8bf*)(p);
    const v8bf hi = *(const v8bf*)(p + 8);
    Bf[j] = __builtin_shufflevector(lo, hi, 0, 1, 2, 3, 4, 5, 6, 7,
                                    8, 9, 10, 11, 12, 13, 14, 15);
  }

  const int n0  = chunk * NWT;
  const int r0g = n0 * STRIDE_;        // first global story row of this chunk

  float rm0 = -INFINITY, rm1 = -INFINITY;  // running max (thread 0 authoritative)
  int m_done = 0;

  // A layout (16x32 bf16): M = lane&15; K = kb8 + elem (kb8 = 16*(lane>=16)).
  const int arow = lane & 15;
  const int kb8  = (lane >= 16) ? 16 : 0;
  const int chi  = (lane >= 16) ? 8 : 0;   // C row offset for this lane half
  const int ccol = wv * 16 + (lane & 15);  // local score column

  __syncthreads();   // s_gw visible

  for (int t = 0; t < NTILE; ++t) {
    // ---- stage 16 normalized l_s rows as bf16 into LDS (2 rows per wave)
#pragma unroll
    for (int rr = 0; rr < 2; ++rr) {
      const int rloc = 2 * wv + rr;
      const float* src = ls + ((size_t)b * S_ + (size_t)(r0g + 16 * t + rloc)) * D_;
      float v[10];
      float ss = 0.f;
#pragma unroll
      for (int i = 0; i < 10; ++i) {
        const int k = lane + 32 * i;
        const float x = (k < D_) ? src[k] : 0.f;
        v[i] = x;
        ss += x * x;
      }
#pragma unroll
      for (int off = 16; off >= 1; off >>= 1) ss += __shfl_xor(ss, off, 32);
      const float inv = rsqrtf(ss + 1e-6f);
#pragma unroll
      for (int i = 0; i < 10; ++i) {
        const int k = lane + 32 * i;
        s_a[rloc * KP + k] = (__bf16)(v[i] * inv);
      }
    }
    __syncthreads();

    // ---- WMMA: one 16x16 f32 tile per wave, K = 320 in 10 steps
    v8f acc = {};
#pragma unroll
    for (int j = 0; j < 10; ++j) {
      const __bf16* p = &s_a[arow * KP + 32 * j + kb8];
      const v8bf lo = *(const v8bf*)(p);
      const v8bf hi = *(const v8bf*)(p + 8);
      const v16bf Af = __builtin_shufflevector(lo, hi, 0, 1, 2, 3, 4, 5, 6, 7,
                                               8, 9, 10, 11, 12, 13, 14, 15);
      acc = __builtin_amdgcn_wmma_f32_16x16x32_bf16(
          /*neg_a=*/false, Af, /*neg_b=*/false, Bf[j],
          /*c_mod=*/(short)0, acc, /*reuse_a=*/false, /*reuse_b=*/false);
    }

    // ---- scatter C tile into the circular score buffer
    {
      const int rbase = ((16 * t) % RB) + chi;     // no intra-tile wrap (RB%16==0)
#pragma unroll
      for (int vv = 0; vv < 8; ++vv)
        s_sc[(rbase + vv) * COLS + ccol] = acc[vv];
    }
    __syncthreads();

    // ---- pool windows completed by this tile: 5m + 59 <= 16t + 15
    int m_hi = (16 * t - 44) / 5;                  // trunc; negative -> none
    if (m_hi > NWT - 1) m_hi = NWT - 1;
    const int col  = tid & 127;
    const int wsel = tid >> 7;                     // process 2 windows/round
    for (int m0 = m_done; m0 <= m_hi; m0 += 2) {
      const int m = m0 + wsel;
      const bool valid = (m <= m_hi);
      float p = -INFINITY;
      if (valid) {
        int idx = (m * STRIDE_) % RB;
        for (int w = 0; w < W_; ++w) {
          p = fmaxf(p, s_gw[w] * s_sc[idx * COLS + col]);
          idx = (idx + 1 == RB) ? 0 : idx + 1;
        }
      }
      s_redp[tid] = valid ? p : 0.f;
      s_redc[tid] = (valid && p != 0.f) ? 1.f : 0.f;
      __syncthreads();
#pragma unroll
      for (int s = 32; s >= 1; s >>= 1) {          // reduce over L=64 columns
        if ((col & 63) < s) {
          s_redp[tid] += s_redp[tid + s];
          s_redc[tid] += s_redc[tid + s];
        }
        __syncthreads();
      }
      if ((col & 63) == 0) {
        const int g = col >> 6;                    // which of the 2 'a' values
        const float avg = s_redp[tid] / (s_redc[tid] + 1e-5f);
        s_wavg[wsel * 2 + g] = valid ? avg : -INFINITY;
      }
      __syncthreads();
      if (tid == 0) {
        rm0 = fmaxf(rm0, fmaxf(s_wavg[0], s_wavg[2]));
        rm1 = fmaxf(rm1, fmaxf(s_wavg[1], s_wavg[3]));
      }
      __syncthreads();
    }
    if (m_hi >= m_done) m_done = m_hi + 1;
    __syncthreads();   // pooling reads done before next tile overwrites ring/A
  }

  if (tid == 0) {
    const int a0 = half * 2;
    partials[(b * A_ + a0 + 0) * NCHUNK + chunk] = rm0;
    partials[(b * A_ + a0 + 1) * NCHUNK + chunk] = rm1;
  }
}

// ---------------------------------------------------------------------------
// Kernel 3: out[b,a] = alpha * max over chunks.
// ---------------------------------------------------------------------------
__global__ __launch_bounds__(256) void k_final(const float* __restrict__ partials,
                                               const float* __restrict__ alpha,
                                               float* __restrict__ out) {
  const int i = threadIdx.x;   // 0..255 = b*4 + a
  float m = -INFINITY;
#pragma unroll
  for (int c = 0; c < NCHUNK; ++c) m = fmaxf(m, partials[i * NCHUNK + c]);
  out[i] = m * alpha[0];
}

// ---------------------------------------------------------------------------
extern "C" void kernel_launch(void* const* d_in, const int* in_sizes, int n_in,
                              void* d_out, int out_size, void* d_ws, size_t ws_size,
                              hipStream_t stream) {
  (void)in_sizes; (void)n_in; (void)out_size; (void)ws_size;
  const float* ls    = (const float*)d_in[0];   // [64,4096,300]
  const float* la    = (const float*)d_in[1];   // [64,4,64,300]
  const float* alpha = (const float*)d_in[2];   // [1]
  const float* gw    = (const float*)d_in[3];   // [60]
  float* out = (float*)d_out;                   // [64,4]

  const size_t wsa_bytes = (size_t)B_ * 256 * KP * sizeof(__bf16);  // ~10.5 MB
  __bf16* wsa     = (__bf16*)d_ws;
  float* partials = (float*)((char*)d_ws + wsa_bytes);              // [64*4*8]

  // 1) normalize answers -> bf16 B-matrix workspace
  hipLaunchKernelGGL(k_norm_a, dim3((B_ * 256) / 8), dim3(256), 0, stream, la, wsa);
  // 2) fused GEMM + sliding gaussian max + masked mean + window max
  hipLaunchKernelGGL(k_main, dim3(B_, NCHUNK, 2), dim3(256), 0, stream,
                     ls, wsa, gw, partials);
  // 3) final reduce over chunks, scale by alpha
  hipLaunchKernelGGL(k_final, dim3(1), dim3(256), 0, stream, partials, alpha, out);
}
